// GRUAttentionDecoder_88081189306788
// MI455X (gfx1250) — compile-verified, matrix operand order checked by reading, not run on hardware
//
#include <hip/hip_runtime.h>
#include <math.h>

// ---------------------------------------------------------------------------
// GRU attention decoder for MI455X (gfx1250), bf16 WMMA pipeline.
// B=32, T=64, S=256, H=512, E=256, V=32000, NH=8, HD=64, L=2
// ---------------------------------------------------------------------------

typedef unsigned short u16;
typedef __attribute__((ext_vector_type(16))) __bf16 v16bf;
typedef __attribute__((ext_vector_type(8)))  float  v8f;

#define CB 32
#define CT 64
#define CS 256
#define CH 512
#define CE 256
#define CV 32000
#define CNH 8
#define CHD 64

__device__ __forceinline__ u16 f2bf(float f) {
  unsigned u = __float_as_uint(f);
  unsigned r = u + 0x7FFFu + ((u >> 16) & 1u);   // round-to-nearest-even
  return (u16)(r >> 16);
}

// --------------------------- elementwise kernels ---------------------------

__global__ void k_f32_to_bf16(const float* __restrict__ s, u16* __restrict__ d, int n) {
  for (int i = blockIdx.x * blockDim.x + threadIdx.x; i < n; i += gridDim.x * blockDim.x)
    d[i] = f2bf(s[i]);
}

__global__ void k_f32_to_bf16_2d(const float* __restrict__ s, u16* __restrict__ d,
                                 int rows, int cols, int ldd) {
  int i = blockIdx.x * blockDim.x + threadIdx.x;
  if (i >= rows * cols) return;
  int r = i / cols, c = i - r * cols;
  d[r * ldd + c] = f2bf(s[i]);
}

__global__ void k_init_hidden(const float* __restrict__ dec,
                              float* __restrict__ hidF, u16* __restrict__ hidB) {
  int i = blockIdx.x * blockDim.x + threadIdx.x;
  if (i >= 2 * CB * CH) return;                    // [layer][b][j]
  int b = (i >> 9) & 31, j = i & 511;
  float v = fminf(fmaxf(dec[b * CH + j], -10.f), 10.f);
  hidF[i] = v; hidB[i] = f2bf(v);
}

__global__ void k_zero_out0(float* __restrict__ out) {
  int i = blockIdx.x * blockDim.x + threadIdx.x;
  if (i >= CB * CV) return;
  int b = i / CV, v = i - b * CV;
  out[(size_t)b * CT * CV + v] = 0.f;              // outputs[:, 0, :] = 0
}

__global__ void k_gather_embed(const int* __restrict__ tgt, const float* __restrict__ emb,
                               u16* __restrict__ xe, int tcol) {
  int i = blockIdx.x * blockDim.x + threadIdx.x;
  if (i >= CB * CE) return;
  int b = i >> 8, j = i & 255;
  int tok = tgt[b * CT + tcol];
  xe[i] = f2bf(emb[(size_t)tok * CE + j]);
}

// --------------------------- WMMA GEMM: out = act(A @ W^T + bias) ----------
// A: [M x K] bf16 row-major (lda elems). W: [N x K] bf16 row-major (ldw elems).
// 8 waves/block; each wave owns one 16x16 D tile. All dims multiples of 16,
// K multiple of 32. mode: 0=none, 1=clip(+-100), 2=relu.

union FragBF { v16bf v; uint4 q[2]; };

__global__ void wmma_gemm_bf16(const u16* __restrict__ A, int lda,
                               const u16* __restrict__ W, int ldw,
                               const float* __restrict__ bias,
                               float* __restrict__ outF, size_t ldoF,
                               u16* __restrict__ outB, int ldoB,
                               int K, int ntiles, int mode) {
  int lane  = threadIdx.x & 31;
  int wave  = threadIdx.x >> 5;
  int ntile = blockIdx.x * (blockDim.x >> 5) + wave;
  if (ntile >= ntiles) return;                     // wave-uniform: EXEC all-1s for WMMA
  int mtile = blockIdx.y;
  int half  = lane >> 4;                           // K half: 0 -> K{0..7,16..23}, 1 -> K{8..15,24..31}
  int l15   = lane & 15;

  const u16* arow = A + (size_t)(mtile * 16 + l15) * lda + half * 8;
  const u16* wrow = W + (size_t)(ntile * 16 + l15) * ldw + half * 8;

  v8f acc = {};
  for (int k = 0; k < K; k += 32) {
    FragBF a, b;
    a.q[0] = *(const uint4*)(arow + k);
    a.q[1] = *(const uint4*)(arow + k + 16);
    b.q[0] = *(const uint4*)(wrow + k);
    b.q[1] = *(const uint4*)(wrow + k + 16);
    if (k + 64 < K) __builtin_prefetch(wrow + k + 64, 0, 1);  // global_prefetch_b8 on streamed W
    acc = __builtin_amdgcn_wmma_f32_16x16x32_bf16(false, a.v, false, b.v,
                                                  (short)0, acc, false, false);
  }

  int n = ntile * 16 + l15;
  float bv = bias ? bias[n] : 0.f;
#pragma unroll
  for (int r = 0; r < 8; ++r) {
    float v = acc[r] + bv;
    if (mode == 1)      v = fminf(fmaxf(v, -100.f), 100.f);
    else if (mode == 2) v = fmaxf(v, 0.f);
    int m = mtile * 16 + r + half * 8;             // D layout: lanes 16-31 hold M=8..15
    if (outF) outF[(size_t)m * ldoF + n] = v;
    if (outB) outB[(size_t)m * ldoB + n] = f2bf(v);
  }
}

// --------------------------- attention (per b,h) ---------------------------
// 256 blocks (B*NH), 256 threads; thread s computes one score, softmax in LDS,
// then ctx via 4-way partial sums over S.

__global__ void k_attention(const float* __restrict__ q, const float* __restrict__ Kh,
                            const float* __restrict__ Vh, const int* __restrict__ mask,
                            u16* __restrict__ ctxB) {
  __shared__ float sc[CS];
  __shared__ float tmp[CS];
  int b = blockIdx.x >> 3, h = blockIdx.x & 7;
  int tid = threadIdx.x;

  const float* qp = q + b * CH + h * CHD;
  const float* kp = Kh + ((size_t)(b * CS + tid)) * CH + h * CHD;
  float s = 0.f;
#pragma unroll 8
  for (int d = 0; d < CHD; ++d) s = fmaf(qp[d], kp[d], s);
  s *= 0.125f;                                     // 1/sqrt(64)
  if (mask[b * CS + tid] == 0) s = -10000.f;
  sc[tid] = s; tmp[tid] = s;
  __syncthreads();
  for (int st = 128; st > 0; st >>= 1) {
    if (tid < st) tmp[tid] = fmaxf(tmp[tid], tmp[tid + st]);
    __syncthreads();
  }
  float mx = tmp[0];
  __syncthreads();
  float e = __expf(sc[tid] - mx);
  sc[tid] = e; tmp[tid] = e;
  __syncthreads();
  for (int st = 128; st > 0; st >>= 1) {
    if (tid < st) tmp[tid] += tmp[tid + st];
    __syncthreads();
  }
  float inv = 1.0f / tmp[0];
  __syncthreads();

  int d = tid & 63, part = tid >> 6;
  const float* vp = Vh + ((size_t)(b * CS + part * 64)) * CH + h * CHD + d;
  float acc = 0.f;
  for (int s2 = 0; s2 < 64; ++s2) acc = fmaf(sc[part * 64 + s2], vp[(size_t)s2 * CH], acc);
  tmp[tid] = acc;
  __syncthreads();
  if (tid < 64) {
    float tot = (tmp[d] + tmp[64 + d] + tmp[128 + d] + tmp[192 + d]) * inv;
    ctxB[b * CH + h * CHD + d] = f2bf(tot);
  }
}

// --------------------------- GRU gate combine ------------------------------
// gi/gh: [32 x 1536] (r|z|n). Writes pre-clip h (bf16, ld=ldPre) for the next
// GEMM and clipped h (f32 + bf16) as next-step hidden.

__global__ void k_gru_combine(const float* __restrict__ gi, const float* __restrict__ gh,
                              const float* __restrict__ hprev,
                              u16* __restrict__ hpreB, int ldPre,
                              float* __restrict__ hnF, u16* __restrict__ hnB) {
  int i = blockIdx.x * blockDim.x + threadIdx.x;
  if (i >= CB * CH) return;
  int b = i >> 9, j = i & 511;
  const float* gib = gi + b * 3 * CH;
  const float* ghb = gh + b * 3 * CH;
  float r = 1.f / (1.f + __expf(-(gib[j] + ghb[j])));
  float z = 1.f / (1.f + __expf(-(gib[CH + j] + ghb[CH + j])));
  float n = tanhf(gib[2 * CH + j] + r * ghb[2 * CH + j]);
  float hp = (1.f - z) * n + z * hprev[i];
  hpreB[b * ldPre + j] = f2bf(hp);
  float c = fminf(fmaxf(hp, -10.f), 10.f);
  hnF[i] = c; hnB[i] = f2bf(c);
}

// ---------------------------------------------------------------------------

extern "C" void kernel_launch(void* const* d_in, const int* in_sizes, int n_in,
                              void* d_out, int out_size, void* d_ws, size_t ws_size,
                              hipStream_t stream) {
  (void)in_sizes; (void)n_in; (void)out_size; (void)ws_size;
  const int*   tgt      = (const int*)  d_in[0];
  const float* enc      = (const float*)d_in[1];
  const float* dec_init = (const float*)d_in[2];
  const int*   src_mask = (const int*)  d_in[3];
  const float* emb_tab  = (const float*)d_in[4];
  const float* w_ep = (const float*)d_in[5];  const float* b_ep = (const float*)d_in[6];
  const float* wq   = (const float*)d_in[7];  const float* bq   = (const float*)d_in[8];
  const float* wk   = (const float*)d_in[9];  const float* bk   = (const float*)d_in[10];
  const float* wv   = (const float*)d_in[11]; const float* bv   = (const float*)d_in[12];
  const float* wo   = (const float*)d_in[13]; const float* bo   = (const float*)d_in[14];
  const float* w_ih0 = (const float*)d_in[15]; const float* w_hh0 = (const float*)d_in[16];
  const float* b_ih0 = (const float*)d_in[17]; const float* b_hh0 = (const float*)d_in[18];
  const float* w_ih1 = (const float*)d_in[19]; const float* w_hh1 = (const float*)d_in[20];
  const float* b_ih1 = (const float*)d_in[21]; const float* b_hh1 = (const float*)d_in[22];
  const float* w_o1 = (const float*)d_in[23]; const float* b_o1 = (const float*)d_in[24];
  const float* w_o2 = (const float*)d_in[25]; const float* b_o2 = (const float*)d_in[26];
  float* out = (float*)d_out;

  // ---- workspace carve-up (256B aligned) ----
  char* wbase = (char*)d_ws;
  size_t off = 0;
  auto wsa = [&](size_t bytes) -> char* {
    off = (off + 255) & ~(size_t)255;
    char* p = wbase + off; off += bytes; return p;
  };
  u16* wep_b  = (u16*)wsa((size_t)CH * CE * 2);
  u16* wq_b   = (u16*)wsa((size_t)CH * CH * 2);
  u16* wk_b   = (u16*)wsa((size_t)CH * CH * 2);
  u16* wv_b   = (u16*)wsa((size_t)CH * CH * 2);
  u16* wo_b   = (u16*)wsa((size_t)CH * CH * 2);
  u16* wih0_b = (u16*)wsa((size_t)3 * CH * 2 * CH * 2);
  u16* whh0_b = (u16*)wsa((size_t)3 * CH * CH * 2);
  u16* wih1_b = (u16*)wsa((size_t)3 * CH * CH * 2);
  u16* whh1_b = (u16*)wsa((size_t)3 * CH * CH * 2);
  u16* wo1_b  = (u16*)wsa((size_t)CH * 2 * CH * 2);
  u16* wo2_b  = (u16*)wsa((size_t)CV * CH * 2);
  u16* enc_b  = (u16*)wsa((size_t)CB * CS * CH * 2);
  float* Kh   = (float*)wsa((size_t)CB * CS * CH * 4);
  float* Vh   = (float*)wsa((size_t)CB * CS * CH * 4);
  float* hidF = (float*)wsa((size_t)2 * 2 * CB * CH * 4);  // [buf][layer][b][j]
  u16*   hidB = (u16*)  wsa((size_t)2 * 2 * CB * CH * 2);
  u16* xe_b   = (u16*)wsa((size_t)CB * CE * 2);
  u16* x_b    = (u16*)wsa((size_t)CB * 2 * CH * 2);        // [emb | ctx] bf16
  float* qF   = (float*)wsa((size_t)CB * CH * 4);
  u16* ctx_b  = (u16*)wsa((size_t)CB * CH * 2);
  float* ctxoF= (float*)wsa((size_t)CB * CH * 4);
  float* gi0F = (float*)wsa((size_t)CB * 3 * CH * 4);
  float* gh0F = (float*)wsa((size_t)CB * 3 * CH * 4);
  float* gi1F = (float*)wsa((size_t)CB * 3 * CH * 4);
  float* gh1F = (float*)wsa((size_t)CB * 3 * CH * 4);
  u16* h0pre_b= (u16*)wsa((size_t)CB * CH * 2);
  u16* comb_b = (u16*)wsa((size_t)CB * 2 * CH * 2);        // [h1 | ctx] bf16
  u16* r1_b   = (u16*)wsa((size_t)CB * CH * 2);

  auto conv = [&](const float* s, u16* d, size_t n) {
    k_f32_to_bf16<<<dim3((unsigned)((n + 255) / 256)), dim3(256), 0, stream>>>(s, d, (int)n);
  };
  auto gemm = [&](const u16* A, int lda, const u16* W, int ldw, const float* bias,
                  float* oF, size_t ldoF, u16* oB, int ldoB,
                  int M, int N, int K, int mode) {
    int ntiles = N / 16;
    dim3 grid((unsigned)((ntiles + 7) / 8), (unsigned)(M / 16));
    wmma_gemm_bf16<<<grid, dim3(256), 0, stream>>>(A, lda, W, ldw, bias,
                                                   oF, ldoF, oB, ldoB, K, ntiles, mode);
  };

  // ---- prologue: weight conversion + step-invariant K/V projections ----
  conv(w_ep, wep_b, (size_t)CH * CE);
  conv(wq, wq_b, (size_t)CH * CH);
  conv(wk, wk_b, (size_t)CH * CH);
  conv(wv, wv_b, (size_t)CH * CH);
  conv(wo, wo_b, (size_t)CH * CH);
  conv(w_ih0, wih0_b, (size_t)3 * CH * 2 * CH);
  conv(w_hh0, whh0_b, (size_t)3 * CH * CH);
  conv(w_ih1, wih1_b, (size_t)3 * CH * CH);
  conv(w_hh1, whh1_b, (size_t)3 * CH * CH);
  conv(w_o1, wo1_b, (size_t)CH * 2 * CH);
  conv(w_o2, wo2_b, (size_t)CV * CH);
  conv(enc, enc_b, (size_t)CB * CS * CH);

  // Kh = clip(enc @ wk^T + bk, +-100); Vh = enc @ wv^T + bv   (M = B*S = 8192)
  gemm(enc_b, CH, wk_b, CH, bk, Kh, CH, nullptr, 0, CB * CS, CH, CH, 1);
  gemm(enc_b, CH, wv_b, CH, bv, Vh, CH, nullptr, 0, CB * CS, CH, CH, 0);

  k_init_hidden<<<dim3((2 * CB * CH + 255) / 256), dim3(256), 0, stream>>>(dec_init, hidF, hidB);
  k_zero_out0<<<dim3((CB * CV + 255) / 256), dim3(256), 0, stream>>>(out);

  // ---- decode loop: t = 1..63, input token tgt[:, t-1] ----
  for (int t = 1; t < CT; ++t) {
    int cur = (t - 1) & 1, nxt = t & 1;
    float* hFc = hidF + (size_t)cur * 2 * CB * CH;
    float* hFn = hidF + (size_t)nxt * 2 * CB * CH;
    u16*   hBc = hidB + (size_t)cur * 2 * CB * CH;
    u16*   hBn = hidB + (size_t)nxt * 2 * CB * CH;
    float* hFc0 = hFc;            float* hFc1 = hFc + CB * CH;
    u16*   hBc0 = hBc;            u16*   hBc1 = hBc + CB * CH;
    float* hFn0 = hFn;            float* hFn1 = hFn + CB * CH;
    u16*   hBn0 = hBn;            u16*   hBn1 = hBn + CB * CH;

    // embedding gather + projection -> x[:, 0:512] (bf16)
    k_gather_embed<<<dim3((CB * CE + 255) / 256), dim3(256), 0, stream>>>(tgt, emb_tab, xe_b, t - 1);
    gemm(xe_b, CE, wep_b, CE, b_ep, nullptr, 0, x_b, 2 * CH, CB, CH, CE, 0);

    // q = clip(h1 @ wq^T + bq)
    gemm(hBc1, CH, wq_b, CH, bq, qF, CH, nullptr, 0, CB, CH, CH, 1);

    // attention -> ctx (bf16)
    k_attention<<<dim3(CB * CNH), dim3(256), 0, stream>>>(qF, Kh, Vh, src_mask, ctx_b);

    // ctxo = clip(ctx @ wo^T + bo) -> f32 (for comb) + x[:, 512:1024] (bf16)
    gemm(ctx_b, CH, wo_b, CH, bo, ctxoF, CH, x_b + CH, 2 * CH, CB, CH, CH, 1);

    // GRU layer 0
    gemm(x_b, 2 * CH, wih0_b, 2 * CH, b_ih0, gi0F, 3 * CH, nullptr, 0, CB, 3 * CH, 2 * CH, 0);
    gemm(hBc0, CH, whh0_b, CH, b_hh0, gh0F, 3 * CH, nullptr, 0, CB, 3 * CH, CH, 0);
    k_gru_combine<<<dim3((CB * CH + 255) / 256), dim3(256), 0, stream>>>(
        gi0F, gh0F, hFc0, h0pre_b, CH, hFn0, hBn0);

    // GRU layer 1 (input = pre-clip h0)
    gemm(h0pre_b, CH, wih1_b, CH, b_ih1, gi1F, 3 * CH, nullptr, 0, CB, 3 * CH, CH, 0);
    gemm(hBc1, CH, whh1_b, CH, b_hh1, gh1F, 3 * CH, nullptr, 0, CB, 3 * CH, CH, 0);
    // pre-clip h1 -> comb[:, 0:512]; clipped h1 -> next hidden
    k_gru_combine<<<dim3((CB * CH + 255) / 256), dim3(256), 0, stream>>>(
        gi1F, gh1F, hFc1, comb_b, 2 * CH, hFn1, hBn1);
    // comb[:, 512:1024] = ctxo
    k_f32_to_bf16_2d<<<dim3((CB * CH + 255) / 256), dim3(256), 0, stream>>>(
        ctxoF, comb_b + CH, CB, CH, 2 * CH);

    // output MLP: relu(comb @ w_o1^T + b_o1) then vocab projection
    gemm(comb_b, 2 * CH, wo1_b, 2 * CH, b_o1, nullptr, 0, r1_b, CH, CB, CH, 2 * CH, 2);
    gemm(r1_b, CH, wo2_b, CH, b_o2, out + (size_t)t * CV, (size_t)CT * CV,
         nullptr, 0, CB, CV, CH, 1);
  }
}
